// DPSNLayer_40853728920362
// MI455X (gfx1250) — matrix operand order repaired and checked by reading, not compiled, not run on hardware
//
#include <hip/hip_runtime.h>
#include <hip/hip_bf16.h>

// ---------------------------------------------------------------------------
// DPSN layer: router GEMM (WMMA bf16, 64-token async-LDS A panel, 4x M-reuse
// of B fragments, transposed-W) -> LDS-staged top-k softmax -> aux loss ->
// gather/combine.  Shapes: T=4096 tokens, D=1024, N=16384 slots, k=32
// ---------------------------------------------------------------------------

#define TT 4096      // B*S tokens
#define DD 1024      // feature dim (K of router GEMM)
#define NN 16384     // slots (N of router GEMM)
#define TOPK 32
#define MTILE 64     // tokens per block in router GEMM
#define NGRP 4       // N split across gridDim.y

typedef __attribute__((ext_vector_type(16))) __bf16 v16bf;
typedef __attribute__((ext_vector_type(8)))  float  v8f;
typedef __attribute__((ext_vector_type(4)))  unsigned int v4u;

union Frag16 { v4u q[2]; v16bf v; };

__device__ __forceinline__ unsigned short f32_to_bf16_rne(float f) {
    unsigned int u = __float_as_uint(f);
    unsigned int r = u + 0x7FFFu + ((u >> 16) & 1u);
    return (unsigned short)(r >> 16);
}

// --------------------------- 1a) fp32 -> bf16 (x) --------------------------
__global__ void cvt_bf16_kernel(const float* __restrict__ src,
                                unsigned short* __restrict__ dst, int n) {
    int i = blockIdx.x * blockDim.x + threadIdx.x;
    if (i < n) dst[i] = f32_to_bf16_rne(src[i]);
}

// ---------------- 1b) fp32 [D][N] -> bf16 transposed [N][D] ----------------
__global__ void __launch_bounds__(256)
cvt_transpose_kernel(const float* __restrict__ src,          // [DD][NN]
                     unsigned short* __restrict__ dst) {     // [NN][DD]
    __shared__ unsigned short tile[32][33];
    const int bx = blockIdx.x;            // n tile (NN/32)
    const int by = blockIdx.y;            // d tile (DD/32)
    const int lx = threadIdx.x & 31;
    const int ly = threadIdx.x >> 5;      // 0..7
    #pragma unroll
    for (int r = 0; r < 32; r += 8) {
        const int d = by * 32 + ly + r;
        const int n = bx * 32 + lx;
        tile[ly + r][lx] = f32_to_bf16_rne(src[(size_t)d * NN + n]);
    }
    __syncthreads();
    #pragma unroll
    for (int r = 0; r < 32; r += 8) {
        const int n = bx * 32 + ly + r;
        const int d = by * 32 + lx;
        dst[(size_t)n * DD + d] = tile[lx][ly + r];
    }
}

// --------------------- 2) router GEMM: logits = x @ W ----------------------
// Block = 64 tokens x 4096 columns (gridDim = {64, 4}).  A panel 64x1024 bf16
// = 128 KB dynamic LDS, filled once with global_load_async_to_lds_b128.
// Each wave keeps 4 accumulators (M-subtiles) so every B fragment (32
// contiguous bytes of a W^T column, L2-resident) is reused 4x.  K loop is
// completely barrier-free.
__global__ void __launch_bounds__(256)
router_gemm_kernel(const unsigned short* __restrict__ xbf,    // [TT][DD] bf16
                   const unsigned short* __restrict__ wt,     // [NN][DD] bf16
                   float* __restrict__ logits) {              // [TT][NN] f32
    extern __shared__ __align__(16) unsigned short sA[];      // MTILE*DD bf16

    const int tid  = threadIdx.x;
    const int lane = tid & 31;
    const int wv   = tid >> 5;       // wave id 0..7
    const int h    = lane >> 4;      // lane half 0/1
    const int m16  = lane & 15;
    const int t0   = blockIdx.x * MTILE;
    const int g0   = blockIdx.y * (NN / NGRP);   // first column of this group

    // ---- async-copy the contiguous 128KB A panel into LDS ----
    {
        const unsigned lds_base = (unsigned)(size_t)(&sA[0]);
        const unsigned long long gbase =
            (unsigned long long)(const void*)(xbf + (size_t)t0 * DD);
        #pragma unroll
        for (int it = 0; it < (MTILE * DD * 2) / (256 * 16); ++it) {  // 32
            const unsigned off = (unsigned)(tid + it * 256) * 16u;
            asm volatile("global_load_async_to_lds_b128 %0, %1, off"
                         :: "v"(lds_base + off), "v"(gbase + off) : "memory");
        }
        asm volatile("s_wait_asynccnt 0x0" ::: "memory");
    }
    __syncthreads();

    // ---- each wave sweeps its own 16-column tiles; no further barriers ----
    for (int ct = wv; ct < (NN / NGRP) / 16; ct += 8) {
        const int n0 = g0 + ct * 16;
        const unsigned short* wcol = wt + (size_t)(n0 + m16) * DD;

        v8f acc[4];
        #pragma unroll
        for (int ms = 0; ms < 4; ++ms)
            acc[ms] = (v8f){0.f, 0.f, 0.f, 0.f, 0.f, 0.f, 0.f, 0.f};

        #pragma unroll 2
        for (int kc = 0; kc < DD / 32; ++kc) {
            Frag16 b;
            // B (contiguous k order, same as A): col m16, K = h*16..h*16+15 contiguous
            b.q[0] = *(const v4u*)&wcol[kc * 32 + h * 16];
            b.q[1] = *(const v4u*)&wcol[kc * 32 + h * 16 + 8];
            #pragma unroll
            for (int ms = 0; ms < 4; ++ms) {
                Frag16 a;
                // A (contiguous k order, same as B): row ms*16+m16;
                // K = h*16..h*16+15 contiguous
                const int arow = (ms * 16 + m16) * DD + kc * 32;
                a.q[0] = *(const v4u*)&sA[arow + h * 16];
                a.q[1] = *(const v4u*)&sA[arow + h * 16 + 8];
                acc[ms] = __builtin_amdgcn_wmma_f32_16x16x32_bf16(
                    false, a.v, false, b.v, (short)0, acc[ms], false, false);
            }
        }
        // D tiles: VGPR r -> row ms*16 + r + 8*h, col = lane&15
        #pragma unroll
        for (int ms = 0; ms < 4; ++ms) {
            #pragma unroll
            for (int r = 0; r < 8; ++r) {
                const int mrow = ms * 16 + r + 8 * h;
                logits[(size_t)(t0 + mrow) * NN + n0 + m16] = acc[ms][r];
            }
        }
    }
}

// --------- 3) per-token: max, sum-exp, top-32 + softmax weights ------------
// The 64KB logits row is staged once into dynamic LDS; all subsequent passes
// (sum-exp + 32 argmax rounds) run against LDS.
__global__ void __launch_bounds__(256)
topk_softmax_kernel(const float* __restrict__ logits,
                    float* __restrict__ tmax, float* __restrict__ trz,
                    float* __restrict__ tw, int* __restrict__ tidx) {
    extern __shared__ __align__(16) float srow[];   // NN floats = 64 KB
    const int t   = blockIdx.x;
    const int tid = threadIdx.x;
    const float* row = logits + (size_t)t * NN;

    __shared__ float sval[256];
    __shared__ int   sidx[256];
    __shared__ float selv[TOPK];
    __shared__ int   sel[TOPK];

    const float NEG = -3.402823466e38f;

    // ---- stage row + row max ----
    float lm = NEG;
    for (int i = tid; i < NN; i += 256) {
        const float v = row[i];
        srow[i] = v;
        lm = fmaxf(lm, v);
    }
    sval[tid] = lm; __syncthreads();
    for (int off = 128; off > 0; off >>= 1) {
        if (tid < off) sval[tid] = fmaxf(sval[tid], sval[tid + off]);
        __syncthreads();
    }
    const float rowmax = sval[0];
    __syncthreads();

    // ---- row sum-exp (from LDS) ----
    float ls = 0.f;
    for (int i = tid; i < NN; i += 256) ls += __expf(srow[i] - rowmax);
    sval[tid] = ls; __syncthreads();
    for (int off = 128; off > 0; off >>= 1) {
        if (tid < off) sval[tid] += sval[tid + off];
        __syncthreads();
    }
    if (tid == 0) { tmax[t] = rowmax; trz[t] = 1.0f / sval[0]; }
    __syncthreads();

    // ---- iterative top-32 argmax over LDS (tie-break: lower index) ----
    for (int r = 0; r < TOPK; ++r) {
        float bv = NEG; int bi = 0x7fffffff;
        for (int i = tid; i < NN; i += 256) {
            bool skip = false;
            for (int s = 0; s < r; ++s) skip |= (sel[s] == i);
            if (skip) continue;
            const float v = srow[i];
            if (v > bv || (v == bv && i < bi)) { bv = v; bi = i; }
        }
        sval[tid] = bv; sidx[tid] = bi; __syncthreads();
        for (int off = 128; off > 0; off >>= 1) {
            if (tid < off) {
                float ov = sval[tid + off]; int oi = sidx[tid + off];
                if (ov > sval[tid] || (ov == sval[tid] && oi < sidx[tid])) {
                    sval[tid] = ov; sidx[tid] = oi;
                }
            }
            __syncthreads();
        }
        if (tid == 0) { sel[r] = sidx[0]; selv[r] = sval[0]; }
        __syncthreads();
    }

    // ---- softmax over the 32 selected (selv[0] is the max) ----
    if (tid == 0) {
        const float m = selv[0];
        float s = 0.f;
        for (int k = 0; k < TOPK; ++k) s += __expf(selv[k] - m);
        const float rs = 1.0f / s;
        for (int k = 0; k < TOPK; ++k) {
            tw[t * TOPK + k]   = __expf(selv[k] - m) * rs;
            tidx[t * TOPK + k] = sel[k];
        }
    }
}

// ------------------ 4) mean_probs[n] = mean_t probs[t][n] ------------------
__global__ void mean_probs_kernel(const float* __restrict__ logits,
                                  const float* __restrict__ tmax,
                                  const float* __restrict__ trz,
                                  float* __restrict__ mp) {
    const int n = blockIdx.x * blockDim.x + threadIdx.x;
    if (n >= NN) return;
    float acc = 0.f;
    for (int t = 0; t < TT; ++t) {
        acc += __expf(logits[(size_t)t * NN + n] - tmax[t]) * trz[t];
    }
    mp[n] = acc * (1.0f / (float)TT);
}

// ---------------- 5) aux = N * mean(mean_probs^2) = sum(mp^2) --------------
__global__ void aux_loss_kernel(const float* __restrict__ mp,
                                float* __restrict__ out_aux) {
    __shared__ float red[256];
    const int tid = threadIdx.x;
    float a = 0.f;
    for (int n = tid; n < NN; n += 256) { float v = mp[n]; a += v * v; }
    red[tid] = a; __syncthreads();
    for (int off = 128; off > 0; off >>= 1) {
        if (tid < off) red[tid] += red[tid + off];
        __syncthreads();
    }
    if (tid == 0) out_aux[0] = red[0];
}

// ------------- 6) gather params, proj, weighted combine --------------------
__global__ void __launch_bounds__(256)
combine_kernel(const float* __restrict__ x, const float* __restrict__ pool,
               const float* __restrict__ tw, const int* __restrict__ tidx,
               float* __restrict__ out) {
    const int t   = blockIdx.x;
    const int tid = threadIdx.x;
    __shared__ float sx[DD];
    __shared__ float red[256];

    for (int j = tid; j < DD; j += 256) sx[j] = x[(size_t)t * DD + j];
    __syncthreads();

    float accd[4] = {0.f, 0.f, 0.f, 0.f};
    for (int k = 0; k < TOPK; ++k) {
        const int idx = tidx[t * TOPK + k];
        const float* p = pool + (size_t)idx * DD;
        float pv[4];
        float part = 0.f;
        #pragma unroll
        for (int j = 0; j < 4; ++j) {
            const int d = tid + j * 256;
            const float pe = p[d];
            pv[j] = pe;
            part += sx[d] * pe;
        }
        red[tid] = part; __syncthreads();
        for (int off = 128; off > 0; off >>= 1) {
            if (tid < off) red[tid] += red[tid + off];
            __syncthreads();
        }
        const float dot = red[0];
        __syncthreads();                       // protect red[] before reuse
        const float scale = tw[t * TOPK + k] * dot;
        #pragma unroll
        for (int j = 0; j < 4; ++j) accd[j] += scale * pv[j];
    }
    #pragma unroll
    for (int j = 0; j < 4; ++j) out[(size_t)t * DD + tid + j * 256] = accd[j];
}

// ---------------------------------------------------------------------------
extern "C" void kernel_launch(void* const* d_in, const int* in_sizes, int n_in,
                              void* d_out, int out_size, void* d_ws, size_t ws_size,
                              hipStream_t stream) {
    const float* x    = (const float*)d_in[0];   // [2,2048,1024]
    const float* pool = (const float*)d_in[1];   // [16384,1024]
    const float* wr   = (const float*)d_in[2];   // [1024,16384]
    // d_in[3] = max_k (fixed 32)

    // workspace layout (bytes)
    const size_t O_XBF = 0;                                   //  8 MB bf16 x
    const size_t O_WT  = O_XBF + (size_t)TT * DD * 2;         // 32 MB bf16 W^T
    const size_t O_LOG = O_WT  + (size_t)NN * DD * 2;         // 256 MB logits
    const size_t O_MAX = O_LOG + (size_t)TT * NN * 4;
    const size_t O_RZ  = O_MAX + (size_t)TT * 4;
    const size_t O_TW  = O_RZ  + (size_t)TT * 4;
    const size_t O_TI  = O_TW  + (size_t)TT * TOPK * 4;
    const size_t O_MP  = O_TI  + (size_t)TT * TOPK * 4;

    char* ws = (char*)d_ws;
    unsigned short* xbf    = (unsigned short*)(ws + O_XBF);
    unsigned short* wt     = (unsigned short*)(ws + O_WT);
    float*          logits = (float*)(ws + O_LOG);
    float*          tmax   = (float*)(ws + O_MAX);
    float*          trz    = (float*)(ws + O_RZ);
    float*          tw     = (float*)(ws + O_TW);
    int*            tidx   = (int*)(ws + O_TI);
    float*          mp     = (float*)(ws + O_MP);

    float* out_main = (float*)d_out;                    // [T][D]
    float* out_aux  = out_main + (size_t)TT * DD;       // scalar

    // allow >64KB dynamic LDS for the GEMM (128 KB) and topk (64 KB) kernels
    const int gemmLds = MTILE * DD * 2;                 // 131072 bytes
    const int topkLds = NN * 4;                         // 65536 bytes
    hipFuncSetAttribute((const void*)router_gemm_kernel,
                        hipFuncAttributeMaxDynamicSharedMemorySize, gemmLds);
    hipFuncSetAttribute((const void*)topk_softmax_kernel,
                        hipFuncAttributeMaxDynamicSharedMemorySize, topkLds);

    // 1) convert x to bf16; convert + transpose W to bf16 [N][D]
    cvt_bf16_kernel<<<(TT * DD + 255) / 256, 256, 0, stream>>>(x, xbf, TT * DD);
    {
        dim3 g(NN / 32, DD / 32);
        cvt_transpose_kernel<<<g, 256, 0, stream>>>(wr, wt);
    }
    // 2) router GEMM via WMMA bf16
    {
        dim3 g(TT / MTILE, NGRP);
        router_gemm_kernel<<<g, 256, gemmLds, stream>>>(xbf, wt, logits);
    }
    // 3) per-token softmax stats + top-32 + weights (LDS-staged row)
    topk_softmax_kernel<<<TT, 256, topkLds, stream>>>(logits, tmax, trz, tw, tidx);
    // 4) mean routing probabilities per slot
    mean_probs_kernel<<<NN / 256, 256, 0, stream>>>(logits, tmax, trz, mp);
    // 5) aux loss scalar
    aux_loss_kernel<<<1, 256, 0, stream>>>(mp, out_aux);
    // 6) gather + dual-einsum combine
    combine_kernel<<<TT, 256, 0, stream>>>(x, pool, tw, tidx, out_main);
}
